// DiffusionTransformer_17738214932657
// MI455X (gfx1250) — matrix operand order repaired and checked by reading, not compiled, not run on hardware
//
#include <hip/hip_runtime.h>

#define N_TOK 768
#define CA_ 768
#define CS_ 384
#define CP_ 128
#define NH_ 16
#define DH_ 48
#define DHP 64
#define NB_ 24
#define SB_ 4

typedef __attribute__((ext_vector_type(16))) __bf16 v16bf;
typedef __attribute__((ext_vector_type(8))) float v8f;
typedef __attribute__((ext_vector_type(8))) unsigned int v8u;

__device__ __forceinline__ unsigned short f2bf(float f) {
  unsigned u = __float_as_uint(f);
  u += 0x7fffu + ((u >> 16) & 1u);
  return (unsigned short)(u >> 16);
}
__device__ __forceinline__ float bf2f(unsigned short h) {
  return __uint_as_float(((unsigned)h) << 16);
}
__device__ __forceinline__ float sigmoidf_(float x) { return 1.f / (1.f + __expf(-x)); }

// ISA 16-bit A-matrix 16x32 layout: lane = m + 16*k[3], element = k[2:0] | k[4]<<3
__device__ __forceinline__ int afrag_pos(int m, int k) {
  int hi = (k >> 3) & 1;
  int e  = (k & 7) | (((k >> 4) & 1) << 3);
  return ((m + (hi << 4)) << 4) + e;
}
// ISA 16-bit B-matrix 32x16 layout: lane = n + 16*k[3], element = k[2:0] | k[4]<<3
__device__ __forceinline__ int bfrag_pos(int k, int n) { return afrag_pos(n, k); }

__device__ __forceinline__ v16bf ldsfrag(const unsigned short* p) {
  v8u t = *(const v8u*)p;
  return __builtin_bit_cast(v16bf, t);
}

// Async 16B global->LDS copy (gfx1250 GLOBAL_LOAD_ASYNC_TO_LDS_B128, ASYNCcnt).
// LDS destination VGPR holds the byte offset = low 32 bits of the generic address.
__device__ __forceinline__ void async_copy_b128(unsigned short* ldst,
                                                const unsigned short* gsrc) {
  unsigned l = (unsigned)(unsigned long long)ldst;
  unsigned long long g = (unsigned long long)gsrc;
  asm volatile("global_load_async_to_lds_b128 %0, %1, off" ::"v"(l), "v"(g) : "memory");
}
__device__ __forceinline__ void wait_async0() {
  asm volatile("s_wait_asynccnt 0" ::: "memory");
}

// ---------------------------------------------------------------------------
// Generic bf16 WMMA GEMM (double-buffered LDS): C(MxN)=A(MxK,f32)@B(KxN,f32)
// EPI=0: C = acc + bias     EPI=1: C = C + acc * sigmoid(aux)
// M,N multiples of 128; K multiple of 32.
// ---------------------------------------------------------------------------
template <int EPI>
__global__ __launch_bounds__(256) void gemm_bf16(
    const float* __restrict__ A, int lda, const float* __restrict__ B, int ldb,
    const float* __restrict__ bias, float* __restrict__ C, int ldc,
    const float* __restrict__ aux, int M, int Ncols, int K) {
  __shared__ __align__(32) unsigned short a_s[2][8 * 512];
  __shared__ __align__(32) unsigned short b_s[2][8 * 512];
  const int wid = threadIdx.x >> 5, lane = threadIdx.x & 31;
  const int bm = blockIdx.y * 128, bn = blockIdx.x * 128;
  const int wm = wid >> 1, wn = wid & 1;  // 4x2 waves -> wave tile 32x64

  v8f acc[2][4];
  const v8f zf = {0.f, 0.f, 0.f, 0.f, 0.f, 0.f, 0.f, 0.f};
#pragma unroll
  for (int mi = 0; mi < 2; ++mi)
#pragma unroll
    for (int ni = 0; ni < 4; ++ni) acc[mi][ni] = zf;

  auto stage = [&](int kt, int buf) {
    const int k0 = kt << 5;
    for (int i = 0; i < 16; ++i) {  // A: 128x32
      int j = threadIdx.x + (i << 8);
      int r = j >> 5, kk = j & 31;
      float v = A[(size_t)(bm + r) * lda + k0 + kk];
      a_s[buf][((r >> 4) << 9) + afrag_pos(r & 15, kk)] = f2bf(v);
    }
    for (int i = 0; i < 16; ++i) {  // B: 32x128
      int j = threadIdx.x + (i << 8);
      int r = j >> 7, c = j & 127;
      float v = B[(size_t)(k0 + r) * ldb + bn + c];
      b_s[buf][((c >> 4) << 9) + bfrag_pos(r, c & 15)] = f2bf(v);
    }
  };

  const int nk = K >> 5;
  stage(0, 0);
  __syncthreads();
  for (int kt = 0; kt < nk; ++kt) {
    const int cur = kt & 1;
    if (kt + 1 < nk) stage(kt + 1, cur ^ 1);
    v16bf af0 = ldsfrag(&a_s[cur][((2 * wm + 0) << 9) + lane * 16]);
    v16bf af1 = ldsfrag(&a_s[cur][((2 * wm + 1) << 9) + lane * 16]);
#pragma unroll
    for (int ni = 0; ni < 4; ++ni) {
      v16bf bf = ldsfrag(&b_s[cur][((4 * wn + ni) << 9) + lane * 16]);
      acc[0][ni] = __builtin_amdgcn_wmma_f32_16x16x32_bf16(false, af0, false, bf, (short)0,
                                                           acc[0][ni], false, false);
      acc[1][ni] = __builtin_amdgcn_wmma_f32_16x16x32_bf16(false, af1, false, bf, (short)0,
                                                           acc[1][ni], false, false);
    }
    __syncthreads();
  }
  const int half8 = (lane >> 4) << 3;
#pragma unroll
  for (int mi = 0; mi < 2; ++mi)
#pragma unroll
    for (int ni = 0; ni < 4; ++ni) {
      int col = bn + wn * 64 + ni * 16 + (lane & 15);
      float bv = bias ? bias[col] : 0.f;
#pragma unroll
      for (int e = 0; e < 8; ++e) {
        int row = bm + wm * 32 + mi * 16 + half8 + e;
        size_t idx = (size_t)row * ldc + col;
        float v = acc[mi][ni][e] + bv;
        if (EPI == 0) {
          C[idx] = v;
        } else {
          C[idx] = C[idx] + v * sigmoidf_(aux[idx]);
        }
      }
    }
}

// ---------------------------------------------------------------------------
// Pair bias: pl = LN(pair_cond) @ projw + projb, layout plbuf[c'][q][s] bf16.
// ---------------------------------------------------------------------------
__global__ __launch_bounds__(256) void pair_kernel(
    const float* __restrict__ pc, const float* __restrict__ lnw,
    const float* __restrict__ lnb, const float* __restrict__ projw,
    const float* __restrict__ projb, unsigned short* __restrict__ plbuf) {
  const int rowbase = blockIdx.x * 128;  // flat row in N*N
  __shared__ float smean[128], srstd[128];
  __shared__ __align__(32) unsigned short a_s[8 * 4 * 512];  // [msub][kstep]
  __shared__ __align__(32) unsigned short b_s[4 * 4 * 512];  // [nsub][kstep]
  const int wid = threadIdx.x >> 5, lane = threadIdx.x & 31;

  {  // row stats: 2 threads per row
    int r = threadIdx.x >> 1, hh = threadIdx.x & 1;
    const float* rp = pc + (size_t)(rowbase + r) * CP_ + hh * 64;
    float s1 = 0.f, s2 = 0.f;
    for (int c = 0; c < 64; ++c) { float v = rp[c]; s1 += v; s2 += v * v; }
    s1 += __shfl_xor(s1, 1);
    s2 += __shfl_xor(s2, 1);
    if (hh == 0) {
      float m = s1 * (1.f / 128.f);
      float var = s2 * (1.f / 128.f) - m * m;
      smean[r] = m;
      srstd[r] = rsqrtf(fmaxf(var, 0.f) + 1e-5f);
    }
  }
  __syncthreads();
  for (int i = 0; i < 64; ++i) {  // stage A 128x128 with LN fused
    int j = threadIdx.x + (i << 8);
    int r = j >> 7, c = j & 127;
    float v = (pc[(size_t)(rowbase + r) * CP_ + c] - smean[r]) * srstd[r] * lnw[c] + lnb[c];
    a_s[((r >> 4) << 11) + ((c >> 5) << 9) + afrag_pos(r & 15, c & 31)] = f2bf(v);
  }
  for (int i = 0; i < 32; ++i) {  // stage B 128x64
    int j = threadIdx.x + (i << 8);
    int r = j >> 6, c = j & 63;
    b_s[((c >> 4) << 11) + ((r >> 5) << 9) + bfrag_pos(r & 31, c & 15)] = f2bf(projw[r * 64 + c]);
  }
  __syncthreads();

  v8f acc[4];
  const v8f zf = {0.f, 0.f, 0.f, 0.f, 0.f, 0.f, 0.f, 0.f};
#pragma unroll
  for (int ni = 0; ni < 4; ++ni) acc[ni] = zf;
#pragma unroll
  for (int ks = 0; ks < 4; ++ks) {
    v16bf af = ldsfrag(&a_s[(wid << 11) + (ks << 9) + lane * 16]);
#pragma unroll
    for (int ni = 0; ni < 4; ++ni) {
      v16bf bf = ldsfrag(&b_s[(ni << 11) + (ks << 9) + lane * 16]);
      acc[ni] = __builtin_amdgcn_wmma_f32_16x16x32_bf16(false, af, false, bf, (short)0,
                                                        acc[ni], false, false);
    }
  }
  const int half8 = (lane >> 4) << 3;
#pragma unroll
  for (int ni = 0; ni < 4; ++ni) {
    int cp = ni * 16 + (lane & 15);
    float bv = projb[cp];
#pragma unroll
    for (int e = 0; e < 8; ++e) {
      int flat = rowbase + wid * 16 + half8 + e;
      int q = flat / N_TOK;
      int s = flat - q * N_TOK;
      plbuf[((size_t)cp * N_TOK + q) * N_TOK + s] = f2bf(acc[ni][e] + bv);
    }
  }
}

// ---------------------------------------------------------------------------
// Flash attention: grid (NH, N/128); wave = 16 query rows; K/V chunks of 64.
// qbuf: bf16 [h][qblk(6)][8192]   (A-frag order, q pre-scaled by DH^-0.5)
// kbuf: bf16 [h][chunk(12)][4096] (B-frag order for Q@K^T)
// vbuf: bf16 [h][chunk(12)][4096] (B-frag order for P@V)
// pl: bf16 [16 heads][q][s] (already offset to this block's j)
// ---------------------------------------------------------------------------
__global__ __launch_bounds__(256) void attn_kernel(
    const unsigned short* __restrict__ qbuf, const unsigned short* __restrict__ kbuf,
    const unsigned short* __restrict__ vbuf, const unsigned short* __restrict__ pl,
    const int* __restrict__ mask, const float* __restrict__ gatef,
    float* __restrict__ wabuf) {
  const int h = blockIdx.x;
  const int qblk = blockIdx.y * 128;
  const int wid = threadIdx.x >> 5, lane = threadIdx.x & 31;
  __shared__ __align__(32) unsigned short qs[8 * 1024];
  __shared__ __align__(32) unsigned short ks[8 * 512];
  __shared__ __align__(32) unsigned short vs[8 * 512];
  __shared__ __align__(32) unsigned short ps[8][1024];

  {  // async-stage Q tile (16KB contiguous, already frag-ordered)
    const unsigned short* qsrc = qbuf + ((size_t)(h * 6 + blockIdx.y) << 13);
    for (int i = 0; i < 4; ++i) {
      int t = threadIdx.x + (i << 8);
      async_copy_b128(&qs[t * 8], qsrc + t * 8);
    }
    wait_async0();
  }
  __syncthreads();
  v16bf qf0 = ldsfrag(&qs[(wid << 10) + lane * 16]);
  v16bf qf1 = ldsfrag(&qs[(wid << 10) + 512 + lane * 16]);

  v8f o[4];
  const v8f zf = {0.f, 0.f, 0.f, 0.f, 0.f, 0.f, 0.f, 0.f};
#pragma unroll
  for (int ni = 0; ni < 4; ++ni) o[ni] = zf;
  float rm[8], rl[8];
#pragma unroll
  for (int e = 0; e < 8; ++e) { rm[e] = -1e30f; rl[e] = 0.f; }
  const int half8 = (lane >> 4) << 3;
  const int qrow0 = qblk + wid * 16 + half8;

  for (int kc = 0; kc < N_TOK; kc += 64) {
    {  // async-stage K,V chunks (8KB each, contiguous, frag-ordered)
      const unsigned short* ksrc = kbuf + ((size_t)(h * 12 + (kc >> 6)) << 12);
      const unsigned short* vsrc = vbuf + ((size_t)(h * 12 + (kc >> 6)) << 12);
      for (int i = 0; i < 2; ++i) {
        int t = threadIdx.x + (i << 8);
        async_copy_b128(&ks[t * 8], ksrc + t * 8);
        async_copy_b128(&vs[t * 8], vsrc + t * 8);
      }
      wait_async0();
    }
    __syncthreads();

    float sc[4][8];
#pragma unroll
    for (int t = 0; t < 4; ++t) {
      v8f sacc = zf;
      {
        v16bf kf = ldsfrag(&ks[(0 * 4 + t) * 512 + lane * 16]);
        sacc = __builtin_amdgcn_wmma_f32_16x16x32_bf16(false, qf0, false, kf, (short)0, sacc,
                                                       false, false);
        kf = ldsfrag(&ks[(1 * 4 + t) * 512 + lane * 16]);
        sacc = __builtin_amdgcn_wmma_f32_16x16x32_bf16(false, qf1, false, kf, (short)0, sacc,
                                                       false, false);
      }
      int key = kc + t * 16 + (lane & 15);
      float mb = mask[key] ? 0.f : -1e9f;
#pragma unroll
      for (int e = 0; e < 8; ++e) {
        int q = qrow0 + e;
        float pb = bf2f(pl[((size_t)h * N_TOK + q) * N_TOK + key]);
        sc[t][e] = sacc[e] + mb + pb;
      }
    }
    // online softmax (rows live in 16-lane halves of the wave)
#pragma unroll
    for (int e = 0; e < 8; ++e) {
      float mx = fmaxf(fmaxf(sc[0][e], sc[1][e]), fmaxf(sc[2][e], sc[3][e]));
      mx = fmaxf(mx, __shfl_xor(mx, 1));
      mx = fmaxf(mx, __shfl_xor(mx, 2));
      mx = fmaxf(mx, __shfl_xor(mx, 4));
      mx = fmaxf(mx, __shfl_xor(mx, 8));
      float mn = fmaxf(rm[e], mx);
      float lsum = 0.f;
#pragma unroll
      for (int t = 0; t < 4; ++t) {
        float p = __expf(sc[t][e] - mn);
        sc[t][e] = p;
        lsum += p;
      }
      lsum += __shfl_xor(lsum, 1);
      lsum += __shfl_xor(lsum, 2);
      lsum += __shfl_xor(lsum, 4);
      lsum += __shfl_xor(lsum, 8);
      float corr = __expf(rm[e] - mn);
      rl[e] = rl[e] * corr + lsum;
      rm[e] = mn;
#pragma unroll
      for (int ni = 0; ni < 4; ++ni) o[ni][e] = o[ni][e] * corr;
#pragma unroll
      for (int t = 0; t < 4; ++t) {  // P -> wave-private LDS in A-frag order
        int kl = t * 16 + (lane & 15);
        ps[wid][((kl >> 5) << 9) + afrag_pos(half8 + e, kl & 31)] = f2bf(sc[t][e]);
      }
    }
    __syncthreads();
    v16bf pf0 = ldsfrag(&ps[wid][lane * 16]);
    v16bf pf1 = ldsfrag(&ps[wid][512 + lane * 16]);
#pragma unroll
    for (int ni = 0; ni < 4; ++ni) {
      v16bf vf = ldsfrag(&vs[(0 * 4 + ni) * 512 + lane * 16]);
      o[ni] = __builtin_amdgcn_wmma_f32_16x16x32_bf16(false, pf0, false, vf, (short)0, o[ni],
                                                      false, false);
      vf = ldsfrag(&vs[(1 * 4 + ni) * 512 + lane * 16]);
      o[ni] = __builtin_amdgcn_wmma_f32_16x16x32_bf16(false, pf1, false, vf, (short)0, o[ni],
                                                      false, false);
    }
    __syncthreads();
  }
  float inv[8];
#pragma unroll
  for (int e = 0; e < 8; ++e) inv[e] = 1.f / rl[e];
#pragma unroll
  for (int ni = 0; ni < 3; ++ni) {  // d < 48 only
    int d = ni * 16 + (lane & 15);
#pragma unroll
    for (int e = 0; e < 8; ++e) {
      int q = qrow0 + e;
      size_t idx = (size_t)q * CA_ + h * DH_ + d;
      wabuf[idx] = o[ni][e] * inv[e] * sigmoidf_(gatef[idx]);
    }
  }
}

// --------------------------- elementwise kernels ---------------------------
__global__ __launch_bounds__(256) void ln_row_kernel(const float* __restrict__ in,
                                                     float* __restrict__ out, int width) {
  int row = blockIdx.x;
  const float* p = in + (size_t)row * width;
  float s1 = 0.f, s2 = 0.f;
  for (int c = threadIdx.x; c < width; c += 256) { float v = p[c]; s1 += v; s2 += v * v; }
  __shared__ float r1[8], r2[8];
  for (int d = 16; d > 0; d >>= 1) { s1 += __shfl_xor(s1, d); s2 += __shfl_xor(s2, d); }
  int wid = threadIdx.x >> 5, lane = threadIdx.x & 31;
  if (lane == 0) { r1[wid] = s1; r2[wid] = s2; }
  __syncthreads();
  s1 = 0.f; s2 = 0.f;
  for (int w = 0; w < 8; ++w) { s1 += r1[w]; s2 += r2[w]; }
  float m = s1 / width, var = s2 / width - m * m;
  float rstd = rsqrtf(fmaxf(var, 0.f) + 1e-5f);
  for (int c = threadIdx.x; c < width; c += 256) out[(size_t)row * width + c] = (p[c] - m) * rstd;
}

__global__ __launch_bounds__(256) void adaln_kernel(const float* __restrict__ act,
                                                    const float* __restrict__ S,
                                                    const float* __restrict__ Bb,
                                                    float* __restrict__ x) {
  int row = blockIdx.x;
  const float* p = act + (size_t)row * CA_;
  float s1 = 0.f, s2 = 0.f;
  for (int c = threadIdx.x; c < CA_; c += 256) { float v = p[c]; s1 += v; s2 += v * v; }
  __shared__ float r1[8], r2[8];
  for (int d = 16; d > 0; d >>= 1) { s1 += __shfl_xor(s1, d); s2 += __shfl_xor(s2, d); }
  int wid = threadIdx.x >> 5, lane = threadIdx.x & 31;
  if (lane == 0) { r1[wid] = s1; r2[wid] = s2; }
  __syncthreads();
  s1 = 0.f; s2 = 0.f;
  for (int w = 0; w < 8; ++w) { s1 += r1[w]; s2 += r2[w]; }
  float m = s1 / CA_, var = s2 / CA_ - m * m;
  float rstd = rsqrtf(fmaxf(var, 0.f) + 1e-5f);
  for (int c = threadIdx.x; c < CA_; c += 256) {
    size_t idx = (size_t)row * CA_ + c;
    x[idx] = sigmoidf_(S[idx]) * ((p[c] - m) * rstd) + Bb[idx];
  }
}

__global__ __launch_bounds__(256) void cscale_kernel(const float* __restrict__ lns,
                                                     const float* __restrict__ w,
                                                     float* __restrict__ out) {
  int idx = blockIdx.x * 256 + threadIdx.x;  // N*CS
  out[idx] = lns[idx] * w[idx % CS_];
}

// pack q/k/v into attention-ready, fragment-swizzled bf16 layouts
__global__ __launch_bounds__(256) void pack_q_kernel(const float* __restrict__ src,
                                                     unsigned short* __restrict__ dst,
                                                     float scale) {
  int idx = blockIdx.x * 256 + threadIdx.x;  // NH*N*64
  int h = idx / (N_TOK * DHP);
  int rem = idx - h * (N_TOK * DHP);
  int n = rem >> 6, d = rem & 63;
  float v = (d < DH_) ? src[(size_t)n * CA_ + h * DH_ + d] * scale : 0.f;
  int qb = n >> 7, r = n & 127;
  int pos = ((r >> 4) << 10) + ((d >> 5) << 9) + afrag_pos(r & 15, d & 31);
  dst[((size_t)(h * 6 + qb) << 13) + pos] = f2bf(v);
}

__global__ __launch_bounds__(256) void pack_k_kernel(const float* __restrict__ src,
                                                     unsigned short* __restrict__ dst) {
  int idx = blockIdx.x * 256 + threadIdx.x;
  int h = idx / (N_TOK * DHP);
  int rem = idx - h * (N_TOK * DHP);
  int n = rem >> 6, d = rem & 63;
  float v = (d < DH_) ? src[(size_t)n * CA_ + h * DH_ + d] : 0.f;
  int ch = n >> 6, kk = n & 63;
  int pos = ((((d >> 5) << 2) + (kk >> 4)) << 9) + bfrag_pos(d & 31, kk & 15);
  dst[((size_t)(h * 12 + ch) << 12) + pos] = f2bf(v);
}

__global__ __launch_bounds__(256) void pack_v_kernel(const float* __restrict__ src,
                                                     unsigned short* __restrict__ dst) {
  int idx = blockIdx.x * 256 + threadIdx.x;
  int h = idx / (N_TOK * DHP);
  int rem = idx - h * (N_TOK * DHP);
  int n = rem >> 6, d = rem & 63;
  float v = (d < DH_) ? src[(size_t)n * CA_ + h * DH_ + d] : 0.f;
  int ch = n >> 6, kk = n & 63;
  int pos = ((((kk >> 5) << 2) + (d >> 4)) << 9) + bfrag_pos(kk & 31, d & 15);
  dst[((size_t)(h * 12 + ch) << 12) + pos] = f2bf(v);
}

__global__ __launch_bounds__(256) void swiglu_kernel(const float* __restrict__ hb,
                                                     float* __restrict__ cb) {
  int idx = blockIdx.x * 256 + threadIdx.x;  // N * 1536
  int n = idx / (2 * CA_);
  int m = idx - n * (2 * CA_);
  float a = hb[(size_t)n * 4 * CA_ + m];
  float b = hb[(size_t)n * 4 * CA_ + 2 * CA_ + m];
  cb[idx] = (a * sigmoidf_(a)) * b;
}

// ---------------------------------------------------------------------------
static void run_gemm(hipStream_t st, const float* A, int lda, const float* B, int ldb,
                     const float* bias, float* C, int ldc, const float* aux, int M, int Nc,
                     int K, int epi) {
  dim3 g(Nc / 128, M / 128), b(256);
  if (epi == 0)
    gemm_bf16<0><<<g, b, 0, st>>>(A, lda, B, ldb, bias, C, ldc, aux, M, Nc, K);
  else
    gemm_bf16<1><<<g, b, 0, st>>>(A, lda, B, ldb, bias, C, ldc, aux, M, Nc, K);
}

extern "C" void kernel_launch(void* const* d_in, const int* in_sizes, int n_in, void* d_out,
                              int out_size, void* d_ws, size_t ws_size, hipStream_t stream) {
  const float* act         = (const float*)d_in[0];
  const int*   mask        = (const int*)d_in[1];
  const float* single_cond = (const float*)d_in[2];
  const float* pair_cond   = (const float*)d_in[3];
  const float* pair_ln_w   = (const float*)d_in[4];
  const float* pair_ln_b   = (const float*)d_in[5];
  const float* pair_proj_w = (const float*)d_in[6];
  const float* pair_proj_b = (const float*)d_in[7];
  const float* sa_cln_w    = (const float*)d_in[8];
  const float* sa_scale_w  = (const float*)d_in[9];
  const float* sa_scale_b  = (const float*)d_in[10];
  const float* sa_bias_w   = (const float*)d_in[11];
  const float* q_w         = (const float*)d_in[12];
  const float* q_b         = (const float*)d_in[13];
  const float* k_w         = (const float*)d_in[14];
  const float* v_w         = (const float*)d_in[15];
  const float* gate_w      = (const float*)d_in[16];
  const float* sa_out_w    = (const float*)d_in[17];
  const float* sa_azc_w    = (const float*)d_in[18];
  const float* sa_azc_b    = (const float*)d_in[19];
  const float* tr_cln_w    = (const float*)d_in[20];
  const float* tr_scale_w  = (const float*)d_in[21];
  const float* tr_scale_b  = (const float*)d_in[22];
  const float* tr_bias_w   = (const float*)d_in[23];
  const float* tr1_w       = (const float*)d_in[24];
  const float* tr1_b       = (const float*)d_in[25];
  const float* tr_out_w    = (const float*)d_in[26];
  const float* tr_azc_w    = (const float*)d_in[27];
  const float* tr_azc_b    = (const float*)d_in[28];

  const size_t NC = (size_t)N_TOK * CA_;
  const size_t NS = (size_t)N_TOK * CS_;
  float* wsf   = (float*)d_ws;
  float* LNsc  = wsf;                 // N*CS
  float* csb   = LNsc + NS;           // N*CS
  float* Sbuf  = csb + NS;            // N*CA
  float* Bbuf  = Sbuf + NC;
  float* azcb  = Bbuf + NC;
  float* xbuf  = azcb + NC;
  float* qf    = xbuf + NC;
  float* kf    = qf + NC;
  float* vf    = kf + NC;
  float* gatef = vf + NC;
  float* wab   = gatef + NC;
  float* hbuf  = wab + NC;            // N*4CA
  float* cbuf  = hbuf + 4 * NC;       // N*2CA
  unsigned short* qpk = (unsigned short*)(cbuf + 2 * NC);  // NH*N*64 each
  unsigned short* kpk = qpk + (size_t)NH_ * N_TOK * DHP;
  unsigned short* vpk = kpk + (size_t)NH_ * N_TOK * DHP;
  unsigned short* plb = vpk + (size_t)NH_ * N_TOK * DHP;   // 64*N*N bf16

  float* actb = (float*)d_out;
  hipMemcpyAsync(actb, act, NC * sizeof(float), hipMemcpyDeviceToDevice, stream);
  ln_row_kernel<<<N_TOK, 256, 0, stream>>>(single_cond, LNsc, CS_);

  const int csBlocks = (int)(NS / 256);
  const float qscale = 0.14433756729740643f;  // 48^-0.5

  for (int sb = 0; sb < NB_ / SB_; ++sb) {
    pair_kernel<<<(N_TOK * N_TOK) / 128, 256, 0, stream>>>(
        pair_cond, pair_ln_w, pair_ln_b, pair_proj_w + (size_t)sb * CP_ * (SB_ * NH_),
        pair_proj_b + (size_t)sb * (SB_ * NH_), plb);
    for (int j = 0; j < SB_; ++j) {
      const int i = sb * SB_ + j;
      // ---- self-attention branch ----
      cscale_kernel<<<csBlocks, 256, 0, stream>>>(LNsc, sa_cln_w + (size_t)i * CS_, csb);
      run_gemm(stream, csb, CS_, sa_scale_w + (size_t)i * CS_ * CA_, CA_,
               sa_scale_b + (size_t)i * CA_, Sbuf, CA_, nullptr, N_TOK, CA_, CS_, 0);
      run_gemm(stream, csb, CS_, sa_bias_w + (size_t)i * CS_ * CA_, CA_, nullptr, Bbuf, CA_,
               nullptr, N_TOK, CA_, CS_, 0);
      run_gemm(stream, csb, CS_, sa_azc_w + (size_t)i * CS_ * CA_, CA_,
               sa_azc_b + (size_t)i * CA_, azcb, CA_, nullptr, N_TOK, CA_, CS_, 0);
      adaln_kernel<<<N_TOK, 256, 0, stream>>>(actb, Sbuf, Bbuf, xbuf);
      run_gemm(stream, xbuf, CA_, q_w + (size_t)i * CA_ * CA_, CA_, q_b + (size_t)i * CA_, qf,
               CA_, nullptr, N_TOK, CA_, CA_, 0);
      run_gemm(stream, xbuf, CA_, k_w + (size_t)i * CA_ * CA_, CA_, nullptr, kf, CA_, nullptr,
               N_TOK, CA_, CA_, 0);
      run_gemm(stream, xbuf, CA_, v_w + (size_t)i * CA_ * CA_, CA_, nullptr, vf, CA_, nullptr,
               N_TOK, CA_, CA_, 0);
      run_gemm(stream, xbuf, CA_, gate_w + (size_t)i * CA_ * CA_, CA_, nullptr, gatef, CA_,
               nullptr, N_TOK, CA_, CA_, 0);
      const int pkBlocks = (NH_ * N_TOK * DHP) / 256;
      pack_q_kernel<<<pkBlocks, 256, 0, stream>>>(qf, qpk, qscale);
      pack_k_kernel<<<pkBlocks, 256, 0, stream>>>(kf, kpk);
      pack_v_kernel<<<pkBlocks, 256, 0, stream>>>(vf, vpk);
      attn_kernel<<<dim3(NH_, N_TOK / 128), 256, 0, stream>>>(
          qpk, kpk, vpk, plb + (size_t)(j * NH_) * N_TOK * N_TOK, mask, gatef, wab);
      run_gemm(stream, wab, CA_, sa_out_w + (size_t)i * CA_ * CA_, CA_, nullptr, actb, CA_,
               azcb, N_TOK, CA_, CA_, 1);
      // ---- SwiGLU transition branch ----
      cscale_kernel<<<csBlocks, 256, 0, stream>>>(LNsc, tr_cln_w + (size_t)i * CS_, csb);
      run_gemm(stream, csb, CS_, tr_scale_w + (size_t)i * CS_ * CA_, CA_,
               tr_scale_b + (size_t)i * CA_, Sbuf, CA_, nullptr, N_TOK, CA_, CS_, 0);
      run_gemm(stream, csb, CS_, tr_bias_w + (size_t)i * CS_ * CA_, CA_, nullptr, Bbuf, CA_,
               nullptr, N_TOK, CA_, CS_, 0);
      run_gemm(stream, csb, CS_, tr_azc_w + (size_t)i * CS_ * CA_, CA_,
               tr_azc_b + (size_t)i * CA_, azcb, CA_, nullptr, N_TOK, CA_, CS_, 0);
      adaln_kernel<<<N_TOK, 256, 0, stream>>>(actb, Sbuf, Bbuf, xbuf);
      run_gemm(stream, xbuf, CA_, tr1_w + (size_t)i * CA_ * 4 * CA_, 4 * CA_,
               tr1_b + (size_t)i * 4 * CA_, hbuf, 4 * CA_, nullptr, N_TOK, 4 * CA_, CA_, 0);
      swiglu_kernel<<<(int)((2 * NC) / 256), 256, 0, stream>>>(hbuf, cbuf);
      run_gemm(stream, cbuf, 2 * CA_, tr_out_w + (size_t)i * 2 * CA_ * CA_, CA_, nullptr,
               actb, CA_, azcb, N_TOK, CA_, 2 * CA_, 1);
    }
  }
}
